// CrossNat3D_84224308675130
// MI455X (gfx1250) — matrix operand order repaired and checked by reading, not compiled, not run on hardware
//
#include <hip/hip_runtime.h>

// ---------------------------------------------------------------------------
// CrossNat3D on gfx1250 (MI455X): f16 WMMA GEMMs + union-window WMMA attention
// ---------------------------------------------------------------------------
typedef __attribute__((ext_vector_type(16))) _Float16 v16h;
typedef __attribute__((ext_vector_type(8)))  _Float16 h8;
typedef __attribute__((ext_vector_type(4)))  _Float16 h4;
typedef __attribute__((ext_vector_type(8)))  float    v8f;
typedef __attribute__((ext_vector_type(4)))  float    vf4;

#define NB_B 2
#define NB_T 8
#define NB_H 64
#define NB_W 64
#define NB_C 128
#define NB_NH 4
#define NB_HD 32
#define NB_KT 3
#define NB_KH 7
#define NB_KW 7
#define NB_TOK (NB_B*NB_T*NB_H*NB_W)      // 65536
#define SCALE_F 0.17677669529663689f       // 32^-0.5

__device__ __forceinline__ int clampi(int v, int lo, int hi) {
    return v < lo ? lo : (v > hi ? hi : v);
}

__device__ __forceinline__ v16h cat16(h8 lo, h8 hi) {
    v16h r;
#pragma unroll
    for (int i = 0; i < 8; ++i) { r[i] = lo[i]; r[8 + i] = hi[i]; }
    return r;
}

// ---------------------------------------------------------------------------
// Weight conversion: f32 [K][N] -> f16 transposed [N][K]
// ---------------------------------------------------------------------------
__global__ void cvt_weights(const float* __restrict__ qv_w,
                            const float* __restrict__ k_w,
                            const float* __restrict__ proj_w,
                            _Float16* __restrict__ wqvT,
                            _Float16* __restrict__ wkT,
                            _Float16* __restrict__ wpT) {
    int i = blockIdx.x * blockDim.x + threadIdx.x;
    if (i < 32768) {                       // qv_w: 128 x 256
        int k = i >> 8, n = i & 255;
        wqvT[n * 128 + k] = (_Float16)qv_w[i];
    } else if (i < 49152) {                // k_w: 128 x 128
        int j = i - 32768;
        int k = j >> 7, n = j & 127;
        wkT[n * 128 + k] = (_Float16)k_w[j];
    } else if (i < 65536) {                // proj_w: 128 x 128
        int j = i - 49152;
        int k = j >> 7, n = j & 127;
        wpT[n * 128 + k] = (_Float16)proj_w[j];
    }
}

// ---------------------------------------------------------------------------
// Wave-tile WMMA GEMM: out[M,N] = A[M,128] @ W[128,N] + bias
// Each wave computes a 16x64 tile (4 accumulators): A-fragment reused across
// 4 WMMAs per K-step -> 4x arithmetic intensity on the f32 A stream.
// MODE 0: A=f32 (x), N=256, split -> q (scaled) / v, f16 out
// MODE 1: A=f32 (y), N=128 -> k f16 out
// MODE 2: A=f16 (attn), N=128 -> f32 out (final)
// ---------------------------------------------------------------------------
template <int MODE>
__global__ __launch_bounds__(256) void gemm_wmma(
    const void* __restrict__ Aptr, const _Float16* __restrict__ Wt,
    const float* __restrict__ bias,
    _Float16* __restrict__ o16a, _Float16* __restrict__ o16b,
    float* __restrict__ o32) {
    constexpr int K = 128;
    constexpr int N = (MODE == 0) ? 256 : 128;
    constexpr int GN = N / 64;             // wave-groups along N

    const int wid    = (blockIdx.x * blockDim.x + threadIdx.x) >> 5;
    const int lane   = threadIdx.x & 31;
    const int hf     = lane >> 4;          // wave half (K split)
    const int ln     = lane & 15;
    const int tile_m = wid / GN;
    const int group  = wid % GN;
    const int row    = tile_m * 16 + ln;   // A row held by this lane
    const int ncol0  = group * 64 + ln;    // first B column held by this lane

    v8f acc[4] = {};
#pragma unroll
    for (int k0 = 0; k0 < K; k0 += 32) {
        v16h a;
        if constexpr (MODE == 2) {
            const _Float16* A = (const _Float16*)Aptr + (size_t)row * K;
            a = cat16(*(const h8*)(A + k0 + 8 * hf),
                      *(const h8*)(A + k0 + 16 + 8 * hf));
        } else {
            const float* A = (const float*)Aptr + (size_t)row * K;
            if (k0 + 32 < K) __builtin_prefetch(A + k0 + 32, 0, 3);
            const vf4* p0 = (const vf4*)(A + k0 + 8 * hf);
            const vf4* p1 = (const vf4*)(A + k0 + 16 + 8 * hf);
            vf4 f0 = p0[0], f1 = p0[1], f2 = p1[0], f3 = p1[1];
#pragma unroll
            for (int i = 0; i < 4; ++i) {
                a[i]      = (_Float16)f0[i];
                a[4 + i]  = (_Float16)f1[i];
                a[8 + i]  = (_Float16)f2[i];
                a[12 + i] = (_Float16)f3[i];
            }
        }
#pragma unroll
        for (int g = 0; g < 4; ++g) {
            const _Float16* Wr = Wt + (size_t)(ncol0 + g * 16) * K + k0 + 8 * hf;
            const v16h b = cat16(*(const h8*)Wr, *(const h8*)(Wr + 16));
            acc[g] = __builtin_amdgcn_wmma_f32_16x16x32_f16(
                false, a, false, b, (short)0, acc[g], false, false);
        }
    }

#pragma unroll
    for (int g = 0; g < 4; ++g) {
        const int ncol = ncol0 + g * 16;
        const float bv = bias[ncol];
#pragma unroll
        for (int r = 0; r < 8; ++r) {
            const int m = tile_m * 16 + r + 8 * hf;
            const float val = acc[g][r] + bv;
            if constexpr (MODE == 0) {
                if (ncol < 128) o16a[(size_t)m * 128 + ncol] = (_Float16)(val * SCALE_F);
                else            o16b[(size_t)m * 128 + (ncol - 128)] = (_Float16)val;
            } else if constexpr (MODE == 1) {
                o16a[(size_t)m * 128 + ncol] = (_Float16)val;
            } else {
                o32[(size_t)m * 128 + ncol] = val;
            }
        }
    }
}

// ---------------------------------------------------------------------------
// NA3D attention. Block = one 4x4 (h,w) query patch at (b,t); 4 waves = 4 heads.
// Union key window (<=3*10*10=300, padded to 320) staged in LDS:
//   k_lds  [320][128] f16   (row-major, QK B-fragments are contiguous)
//   vT_lds [128][320] f16   (dim-major, AV B-fragments are contiguous)
//   scores [4][16][320] f32, probs [4][16][320] f16, rpb, ntok, ncrd
// 20 QK WMMA tiles + vectorized softmax + 10x2 AV WMMAs per wave.
// ---------------------------------------------------------------------------
#define SM_K    0
#define SM_VT   81920
#define SM_SC   163840
#define SM_PR   245760
#define SM_RPB  286720
#define SM_NTOK 300240
#define SM_NCRD 301520
#define SM_TOTAL 302800

__global__ __launch_bounds__(128) void na3d_attn(
    const _Float16* __restrict__ qf, const _Float16* __restrict__ kf,
    const _Float16* __restrict__ vf, const float* __restrict__ rpb,
    _Float16* __restrict__ attn_out) {
    const int blk = blockIdx.x;
    const int tw = blk & 15;
    const int th = (blk >> 4) & 15;
    const int t  = (blk >> 8) & 7;
    const int b  = blk >> 11;
    const int h0 = th * 4, w0 = tw * 4;

    const int tid  = threadIdx.x;
    const int wid  = tid >> 5;          // head
    const int lane = tid & 31;
    const int hf   = lane >> 4;
    const int ln   = lane & 15;

    // union of the 16 queries' windows
    const int st     = clampi(t - 1, 0, NB_T - NB_KT);
    const int uh0    = clampi(h0 - 3, 0, NB_H - NB_KH);
    const int uh_cnt = clampi(h0, 0, NB_H - NB_KH) + NB_KH - uh0;   // <=10
    const int uw0    = clampi(w0 - 3, 0, NB_W - NB_KW);
    const int uw_cnt = clampi(w0, 0, NB_W - NB_KW) + NB_KW - uw0;   // <=10
    const int uhw    = uh_cnt * uw_cnt;
    const int cnt    = NB_KT * uhw;                                 // <=300

    extern __shared__ char smem[];
    _Float16* k_lds  = (_Float16*)(smem + SM_K);
    _Float16* vT_lds = (_Float16*)(smem + SM_VT);
    float*    scores = (float*)(smem + SM_SC) + wid * 16 * 320;
    _Float16* probs  = (_Float16*)(smem + SM_PR) + wid * 16 * 320;
    float*    rpb_l  = (float*)(smem + SM_RPB);
    int*      ntok   = (int*)(smem + SM_NTOK);
    int*      ncrd   = (int*)(smem + SM_NCRD);

    // stage rpb (all heads) + neighbor token/coord tables (divisions only here)
    for (int i = tid; i < NB_NH * 5 * 13 * 13; i += 128) rpb_l[i] = rpb[i];
    for (int j = tid; j < 320; j += 128) {
        const int jj  = (j < cnt) ? j : 0;
        const int jt  = jj / uhw;
        const int rem = jj % uhw;
        const int jh  = rem / uw_cnt, jw = rem % uw_cnt;
        const int ta = st + jt, ha = uh0 + jh, wa = uw0 + jw;
        ntok[j] = ((b * NB_T + ta) * NB_H + ha) * NB_W + wa;
        ncrd[j] = (ta << 12) | (ha << 6) | wa;
    }
    __syncthreads();

    // stage k (row-major) and v (transposed) for the union
    for (int idx = tid; idx < 320 * 16; idx += 128) {
        const int j = idx >> 4, c = idx & 15;
        const size_t base = (size_t)ntok[j] * NB_C + c * 8;
        h8 kv = *(const h8*)(kf + base);
        *(h8*)(k_lds + j * NB_C + c * 8) = kv;
        h8 vv = *(const h8*)(vf + base);
#pragma unroll
        for (int i = 0; i < 8; ++i) vT_lds[(c * 8 + i) * 320 + j] = vv[i];
    }
    __syncthreads();

    // q A-fragment (constant across neighbor tiles); q already * SCALE
    const int qh = h0 + (ln >> 2), qw = w0 + (ln & 3);
    const size_t qtok = ((size_t)(b * NB_T + t) * NB_H + qh) * NB_W + qw;
    const _Float16* qrow = qf + qtok * NB_C + wid * NB_HD;
    const v16h aq = cat16(*(const h8*)(qrow + 8 * hf),
                          *(const h8*)(qrow + 16 + 8 * hf));

    // per-row (loop-invariant) query window bounds for this lane's 8 rows
    int qhr8[8], qwr8[8], qsh8[8], qsw8[8];
#pragma unroll
    for (int r = 0; r < 8; ++r) {
        const int mr = r + 8 * hf;
        qhr8[r] = h0 + (mr >> 2);
        qwr8[r] = w0 + (mr & 3);
        qsh8[r] = clampi(qhr8[r] - 3, 0, NB_H - NB_KH);
        qsw8[r] = clampi(qwr8[r] - 3, 0, NB_W - NB_KW);
    }

    // ---- pass A: QK^T tiles -> masked/biased scores in LDS (branch-free) ----
    for (int jt_ = 0; jt_ < 20; ++jt_) {
        const int j = jt_ * 16 + ln;
        const _Float16* kr = k_lds + j * NB_C + wid * NB_HD + 8 * hf;
        const v16h bk = cat16(*(const h8*)kr, *(const h8*)(kr + 16));
        v8f s = {};
        s = __builtin_amdgcn_wmma_f32_16x16x32_f16(
            false, aq, false, bk, (short)0, s, false, false);

        const int pc = ncrd[j];
        const int ta = pc >> 12, ha = (pc >> 6) & 63, wa = pc & 63;
        const bool jvalid = (j < cnt);
        const int baset = (wid * 5 + (ta - t + (NB_KT - 1))) * 169;

#pragma unroll
        for (int r = 0; r < 8; ++r) {
            // clamp so the bias load is always in-bounds -> no exec branching
            const int rh = clampi(ha - qhr8[r] + (NB_KH - 1), 0, 12);
            const int rw = clampi(wa - qwr8[r] + (NB_KW - 1), 0, 12);
            const float bias = rpb_l[baset + rh * 13 + rw];
            const bool valid = jvalid & (ha >= qsh8[r]) & (ha < qsh8[r] + NB_KH) &
                               (wa >= qsw8[r]) & (wa < qsw8[r] + NB_KW);
            scores[(r + 8 * hf) * 320 + j] = valid ? (s[r] + bias) : -1e30f;
        }
    }
    __syncthreads();

    // ---- pass B: vectorized row softmax; row=ln, halves split the 320 cols ----
    {
        float* sr = scores + ln * 320 + hf * 160;
        vf4* s4 = (vf4*)sr;
        float mx = -1e30f;
        for (int j4 = 0; j4 < 40; ++j4) {
            const vf4 v = s4[j4];
            mx = fmaxf(mx, fmaxf(fmaxf(v[0], v[1]), fmaxf(v[2], v[3])));
        }
        mx = fmaxf(mx, __shfl_xor(mx, 16, 32));
        float ssum = 0.f;
        for (int j4 = 0; j4 < 40; ++j4) {
            vf4 v = s4[j4], e;
#pragma unroll
            for (int i = 0; i < 4; ++i) e[i] = __expf(v[i] - mx);
            s4[j4] = e;
            ssum += (e[0] + e[1]) + (e[2] + e[3]);
        }
        ssum += __shfl_xor(ssum, 16, 32);
        const float inv = 1.f / ssum;
        h4* p4 = (h4*)(probs + ln * 320 + hf * 160);
        for (int j4 = 0; j4 < 40; ++j4) {
            const vf4 e = s4[j4];
            h4 p;
#pragma unroll
            for (int i = 0; i < 4; ++i) p[i] = (_Float16)(e[i] * inv);
            p4[j4] = p;
        }
    }
    __syncthreads();

    // ---- pass C: attn @ v, K-chunks of 32 neighbors, HD split 16+16 ----
    v8f o0 = {}, o1 = {};
    for (int c = 0; c < 10; ++c) {
        const int jc = 32 * c;
        const _Float16* pr = probs + ln * 320 + jc + 8 * hf;
        const v16h ap = cat16(*(const h8*)pr, *(const h8*)(pr + 16));
        const _Float16* v0r = vT_lds + (wid * NB_HD + ln) * 320 + jc + 8 * hf;
        const _Float16* v1r = vT_lds + (wid * NB_HD + ln + 16) * 320 + jc + 8 * hf;
        const v16h b0 = cat16(*(const h8*)v0r, *(const h8*)(v0r + 16));
        const v16h b1 = cat16(*(const h8*)v1r, *(const h8*)(v1r + 16));
        o0 = __builtin_amdgcn_wmma_f32_16x16x32_f16(
            false, ap, false, b0, (short)0, o0, false, false);
        o1 = __builtin_amdgcn_wmma_f32_16x16x32_f16(
            false, ap, false, b1, (short)0, o1, false, false);
    }

#pragma unroll
    for (int r = 0; r < 8; ++r) {
        const int mr = r + 8 * hf;
        const int oh = h0 + (mr >> 2), ow = w0 + (mr & 3);
        const size_t tok = ((size_t)(b * NB_T + t) * NB_H + oh) * NB_W + ow;
        attn_out[tok * NB_C + wid * NB_HD + ln]      = (_Float16)o0[r];
        attn_out[tok * NB_C + wid * NB_HD + ln + 16] = (_Float16)o1[r];
    }
}

// ---------------------------------------------------------------------------
// Launch
// ---------------------------------------------------------------------------
extern "C" void kernel_launch(void* const* d_in, const int* in_sizes, int n_in,
                              void* d_out, int out_size, void* d_ws, size_t ws_size,
                              hipStream_t stream) {
    const float* x      = (const float*)d_in[0];
    const float* y      = (const float*)d_in[1];
    const float* qv_w   = (const float*)d_in[2];
    const float* qv_b   = (const float*)d_in[3];
    const float* k_w    = (const float*)d_in[4];
    const float* k_b    = (const float*)d_in[5];
    const float* proj_w = (const float*)d_in[6];
    const float* proj_b = (const float*)d_in[7];
    const float* rpb    = (const float*)d_in[8];

    // workspace layout (all f16): q, k, v, attn (16 MB each) + weights
    _Float16* q_ws = (_Float16*)d_ws;
    _Float16* k_ws = q_ws + (size_t)NB_TOK * NB_C;
    _Float16* v_ws = k_ws + (size_t)NB_TOK * NB_C;
    _Float16* a_ws = v_ws + (size_t)NB_TOK * NB_C;
    _Float16* wqvT = a_ws + (size_t)NB_TOK * NB_C;
    _Float16* wkT  = wqvT + 256 * 128;
    _Float16* wpT  = wkT + 128 * 128;

    cvt_weights<<<256, 256, 0, stream>>>(qv_w, k_w, proj_w, wqvT, wkT, wpT);

    // qv projection: 4096 M-tiles x 4 N-groups = 16384 waves / 8 per block
    gemm_wmma<0><<<2048, 256, 0, stream>>>(x, wqvT, qv_b, q_ws, v_ws, nullptr);
    // k projection: 4096 x 2 = 8192 waves
    gemm_wmma<1><<<1024, 256, 0, stream>>>(y, wkT, k_b, k_ws, nullptr, nullptr);

    // attention: one block per (b,t,4x4 patch), 4 waves = 4 heads, ~303 KB LDS
    na3d_attn<<<4096, 128, SM_TOTAL, stream>>>(q_ws, k_ws, v_ws, rpb, a_ws);

    // output projection -> f32 d_out
    gemm_wmma<2><<<1024, 256, 0, stream>>>(a_ws, wpT, proj_b, nullptr, nullptr,
                                           (float*)d_out);
}
